// HGAT_35347580846241
// MI455X (gfx1250) — compile-verified, matrix operand order checked by reading, not run on hardware
//
#include <hip/hip_runtime.h>
#include <hip/hip_bf16.h>
#include <math.h>

typedef __attribute__((ext_vector_type(16))) __bf16 v16bf;
typedef __attribute__((ext_vector_type(8)))  __bf16 v8bf;
typedef __attribute__((ext_vector_type(8)))  float  v8f;

namespace {
constexpr int cN = 30000, cP = 8, cDM = 128, cE = 1000, cNNZ = 120000, cDIN = 1024;
}

// ---------------- elementwise / conversion ----------------
__global__ void k_f32_to_bf16(const float* __restrict__ in, __bf16* __restrict__ out, long n) {
  long i = (long)blockIdx.x * blockDim.x + threadIdx.x;
  if (i < n) out[i] = (__bf16)in[i];
}

// W[K,C] row-major -> Wt[C,K] bf16 (so B fragments are K-contiguous)
__global__ void k_transpose_bf16(const float* __restrict__ W, __bf16* __restrict__ Wt, int K, int C) {
  long i = (long)blockIdx.x * blockDim.x + threadIdx.x;
  if (i < (long)K * C) {
    int k = (int)(i / C), c = (int)(i % C);
    Wt[(long)c * K + k] = (__bf16)W[i];
  }
}

__global__ void k_fill(float* __restrict__ p, float v, long n) {
  long i = (long)blockIdx.x * blockDim.x + threadIdx.x;
  if (i < n) p[i] = v;
}

// ---------------- bf16 WMMA GEMM: C = act(A[M,K] @ Bt[Nc,K]^T + bias) (+residual) ----------------
__global__ void k_gemm_bf16(const __bf16* __restrict__ A, const __bf16* __restrict__ Bt,
                            const float* __restrict__ bias, const float* __restrict__ residual,
                            float* __restrict__ C, int M, int Nc, int K, int leaky) {
  const int lane  = threadIdx.x & 31;
  const int wave  = blockIdx.x * (blockDim.x >> 5) + (threadIdx.x >> 5);
  const int tilesN = Nc >> 4;
  const int tilesM = (M + 15) >> 4;
  if (wave >= tilesM * tilesN) return;
  const int tM = wave / tilesN, tN = wave - tM * tilesN;

  int row = tM * 16 + (lane & 15);
  if (row > M - 1) row = M - 1;          // clamp loads; EXEC stays all-1s for WMMA
  const int col = tN * 16 + (lane & 15);
  const int kb  = (lane >> 4) << 3;      // ISA 16-bit A/B layout: hi half-wave owns K+8

  const __bf16* arow = A  + (long)row * K;
  const __bf16* brow = Bt + (long)col * K;

  v8f acc = {};
  for (int k0 = 0; k0 < K; k0 += 32) {
    union { v16bf v; v8bf h[2]; } a, b;
    a.h[0] = *(const v8bf*)(arow + k0 + kb);
    a.h[1] = *(const v8bf*)(arow + k0 + 16 + kb);
    b.h[0] = *(const v8bf*)(brow + k0 + kb);
    b.h[1] = *(const v8bf*)(brow + k0 + 16 + kb);
    acc = __builtin_amdgcn_wmma_f32_16x16x32_bf16(false, a.v, false, b.v,
                                                  (short)0, acc, false, false);
  }

  const float bc = bias ? bias[col] : 0.0f;
  const int rbase = tM * 16 + ((lane >> 4) << 3);  // C/D layout: hi half-wave owns M+8
#pragma unroll
  for (int r = 0; r < 8; ++r) {
    const int rr = rbase + r;
    if (rr < M) {
      float v = acc[r] + bc;
      if (leaky) v = v > 0.0f ? v : 0.2f * v;
      const long idx = (long)rr * Nc + col;
      if (residual) v += residual[idx];
      C[idx] = v;
    }
  }
}

// ---------------- BatchNorm (train-mode, biased var) ----------------
__global__ void k_col_stats(const float* __restrict__ X, int R, int C,
                            float* __restrict__ mean, float* __restrict__ var) {
  __shared__ float s1[256], s2[256];
  const int c = blockIdx.x;
  float a = 0.f, b = 0.f;
  for (int r = threadIdx.x; r < R; r += blockDim.x) {
    float v = X[(long)r * C + c];
    a += v; b += v * v;
  }
  s1[threadIdx.x] = a; s2[threadIdx.x] = b;
  __syncthreads();
  for (int s = blockDim.x >> 1; s > 0; s >>= 1) {
    if ((int)threadIdx.x < s) { s1[threadIdx.x] += s1[threadIdx.x + s]; s2[threadIdx.x] += s2[threadIdx.x + s]; }
    __syncthreads();
  }
  if (threadIdx.x == 0) {
    float m = s1[0] / R;
    mean[c] = m;
    var[c]  = s2[0] / R - m * m;
  }
}

__global__ void k_bn_apply(const float* __restrict__ X, const float* __restrict__ mean,
                           const float* __restrict__ var, const float* __restrict__ g,
                           const float* __restrict__ b, float* __restrict__ Y,
                           __bf16* __restrict__ Ybf, long R, int C) {
  long i = (long)blockIdx.x * blockDim.x + threadIdx.x;
  if (i < R * (long)C) {
    int c = (int)(i % C);
    float v = (X[i] - mean[c]) * rsqrtf(var[c] + 1e-5f) * g[c] + b[c];
    Y[i] = v;
    if (Ybf) Ybf[i] = (__bf16)v;
  }
}

// ---------------- sparse / segment ops ----------------
__global__ void k_scatter_he(const float* __restrict__ h, const int* __restrict__ vidx,
                             const int* __restrict__ eidx, float* __restrict__ he, long total) {
  long i = (long)blockIdx.x * blockDim.x + threadIdx.x;
  if (i >= total) return;
  long k = i >> 7; int d = (int)(i & 127);
  atomicAdd(&he[(long)eidx[k] * 128 + d], h[(long)vidx[k] * 128 + d]);
}

__global__ void k_count_deg(const int* __restrict__ vidx, const int* __restrict__ eidx,
                            float* __restrict__ D, float* __restrict__ B, int nnz) {
  int i = blockIdx.x * blockDim.x + threadIdx.x;
  if (i < nnz) {
    atomicAdd(&D[vidx[i]], 1.0f);
    atomicAdd(&B[eidx[i]], 1.0f);
  }
}

__global__ void k_invert(float* __restrict__ x, long n) {
  long i = (long)blockIdx.x * blockDim.x + threadIdx.x;
  if (i < n) { float v = x[i]; x[i] = v > 0.f ? 1.0f / v : 0.0f; }
}

__global__ void k_alpha(const float* __restrict__ xh, const float* __restrict__ eh,
                        const int* __restrict__ vidx, const int* __restrict__ eidx,
                        const float* __restrict__ att, float* __restrict__ alpha,
                        int H, long n) {
  long i = (long)blockIdx.x * blockDim.x + threadIdx.x;
  if (i >= n) return;
  long k = i / H; int h = (int)(i - k * H);
  const float* xr = xh + (long)vidx[k] * H * 128 + (long)h * 128;
  const float* er = eh + (long)eidx[k] * H * 128 + (long)h * 128;
  const float* a1 = att + (long)h * 256;
  float s = 0.f;
#pragma unroll 8
  for (int d = 0; d < 128; ++d) s += xr[d] * a1[d] + er[d] * a1[128 + d];
  alpha[i] = s > 0.f ? s : 0.2f * s;
}

__device__ __forceinline__ void atomicMaxF(float* addr, float val) {
  unsigned int* ua = (unsigned int*)addr;
  unsigned int old = *ua;
  while (__uint_as_float(old) < val) {
    unsigned int assumed = old;
    old = atomicCAS(ua, assumed, __float_as_uint(val));
    if (old == assumed) break;
  }
}

__global__ void k_seg_max(const float* __restrict__ alpha, const int* __restrict__ eidx,
                          float* __restrict__ emax, int H, long n) {
  long i = (long)blockIdx.x * blockDim.x + threadIdx.x;
  if (i >= n) return;
  long k = i / H; int h = (int)(i - k * H);
  atomicMaxF(&emax[(long)eidx[k] * H + h], alpha[i]);
}

__global__ void k_seg_expsum(float* __restrict__ alpha, const int* __restrict__ eidx,
                             const float* __restrict__ emax, float* __restrict__ esum,
                             int H, long n) {
  long i = (long)blockIdx.x * blockDim.x + threadIdx.x;
  if (i >= n) return;
  long k = i / H; int h = (int)(i - k * H);
  float ex = expf(alpha[i] - emax[(long)eidx[k] * H + h]);
  alpha[i] = ex;
  atomicAdd(&esum[(long)eidx[k] * H + h], ex);
}

__global__ void k_seg_norm(float* __restrict__ alpha, const int* __restrict__ eidx,
                           const float* __restrict__ esum, int H, long n) {
  long i = (long)blockIdx.x * blockDim.x + threadIdx.x;
  if (i >= n) return;
  long k = i / H; int h = (int)(i - k * H);
  alpha[i] = alpha[i] / (esum[(long)eidx[k] * H + h] + 1e-16f);
}

// node -> edge: out_e[e,h,:] += Binv[e] * xh[v,h,:] * alpha[k,h]
__global__ void k_m1(const float* __restrict__ xh, const float* __restrict__ alpha,
                     const float* __restrict__ Binv, const int* __restrict__ vidx,
                     const int* __restrict__ eidx, float* __restrict__ out_e,
                     int H, long total) {
  long i = (long)blockIdx.x * blockDim.x + threadIdx.x;
  if (i >= total) return;
  int d = (int)(i & 127); long t = i >> 7; int h = (int)(t % H); long k = t / H;
  int v = vidx[k], e = eidx[k];
  float val = Binv[e] * xh[(long)v * H * 128 + (long)h * 128 + d] * alpha[k * H + h];
  atomicAdd(&out_e[(long)e * H * 128 + (long)h * 128 + d], val);
}

// edge -> node, head-mean folded in: hc[v,:] += (1/H)*Dinv[v]*out_e[e,h,:]*alpha[k,h]
__global__ void k_m2(const float* __restrict__ out_e, const float* __restrict__ alpha,
                     const float* __restrict__ Dinv, const int* __restrict__ vidx,
                     const int* __restrict__ eidx, float* __restrict__ hc,
                     int H, float invH, long total) {
  long i = (long)blockIdx.x * blockDim.x + threadIdx.x;
  if (i >= total) return;
  int d = (int)(i & 127); long t = i >> 7; int h = (int)(t % H); long k = t / H;
  int v = vidx[k], e = eidx[k];
  float val = invH * Dinv[v] * out_e[(long)e * H * 128 + (long)h * 128 + d] * alpha[k * H + h];
  atomicAdd(&hc[(long)v * 128 + d], val);
}

__global__ void k_add3(const float* __restrict__ A, const float* __restrict__ B,
                       const float* __restrict__ bias, float* __restrict__ Y, long n, int C) {
  long i = (long)blockIdx.x * blockDim.x + threadIdx.x;
  if (i < n) { int c = (int)(i % C); Y[i] = A[i] + B[i] + bias[c]; }
}

// ---------------- host side ----------------
static inline long cdivl(long a, long b) { return (a + b - 1) / b; }

static void launch_gemm(const __bf16* A, const __bf16* Bt, const float* bias,
                        const float* resid, float* C, int M, int Nc, int K, int leaky,
                        hipStream_t s) {
  long waves = (long)((M + 15) / 16) * (Nc / 16);
  int blocks = (int)cdivl(waves, 8);
  k_gemm_bf16<<<blocks, 256, 0, s>>>(A, Bt, bias, resid, C, M, Nc, K, leaky);
}

extern "C" void kernel_launch(void* const* d_in, const int* in_sizes, int n_in,
                              void* d_out, int out_size, void* d_ws, size_t ws_size,
                              hipStream_t stream) {
  (void)in_sizes; (void)n_in; (void)out_size; (void)ws_size;
  const int T = 256;

  const float* x      = (const float*)d_in[0];   // [N,P,DM] == [N,DIN]
  const int*   vidx   = (const int*)d_in[1];
  const int*   eidx   = (const int*)d_in[2];
  const float* W1     = (const float*)d_in[4];   // [DIN,DM]
  const float* b1     = (const float*)d_in[5];
  const float* bn1g   = (const float*)d_in[6];
  const float* bn1b   = (const float*)d_in[7];
  const float* h1_lin = (const float*)d_in[8];   // [DM,4*DM]
  const float* h1_att = (const float*)d_in[9];   // [4,2*DM]
  const float* h1_bias= (const float*)d_in[10];
  const float* bn2g   = (const float*)d_in[11];
  const float* bn2b   = (const float*)d_in[12];
  const float* h2_lin = (const float*)d_in[13];  // [DM,DM]
  const float* h2_att = (const float*)d_in[14];  // [1,2*DM]
  const float* h2_bias= (const float*)d_in[15];
  const float* bn3g   = (const float*)d_in[16];
  const float* bn3b   = (const float*)d_in[17];
  const float* W2     = (const float*)d_in[18];  // [DM,DIN]
  const float* b2     = (const float*)d_in[19];
  const float* bn4g   = (const float*)d_in[20];
  const float* bn4b   = (const float*)d_in[21];
  float* out = (float*)d_out;                    // [N,P,DM] f32

  // bump allocator on workspace (all blocks 256B aligned)
  char* wp = (char*)d_ws;
  auto alloc = [&](size_t bytes) -> char* {
    char* r = wp; wp += (bytes + 255) & ~(size_t)255; return r;
  };
  __bf16* x_bf  = (__bf16*)alloc((size_t)cN * cDIN * 2);
  __bf16* w1t   = (__bf16*)alloc((size_t)cDM * cDIN * 2);       // [128,1024]
  __bf16* h1lt  = (__bf16*)alloc((size_t)512 * cDM * 2);        // [512,128]
  __bf16* h2lt  = (__bf16*)alloc((size_t)cDM * cDM * 2);        // [128,128]
  __bf16* w2t   = (__bf16*)alloc((size_t)cDIN * cDM * 2);       // [1024,128]
  float*  hA    = (float*)alloc((size_t)cN * cDM * 4);
  float*  hB    = (float*)alloc((size_t)cN * cDM * 4);
  float*  hC    = (float*)alloc((size_t)cN * cDM * 4);
  __bf16* hbf   = (__bf16*)alloc((size_t)cN * cDM * 2);
  float*  stats = (float*)alloc((size_t)2 * cDM * 4);
  float*  mean  = stats, *var = stats + cDM;
  float*  he    = (float*)alloc((size_t)cE * cDM * 4);
  __bf16* hebf  = (__bf16*)alloc((size_t)cE * cDM * 2);
  float*  xh    = (float*)alloc((size_t)cN * 4 * cDM * 4);
  float*  eh    = (float*)alloc((size_t)cE * 4 * cDM * 4);
  float*  oute  = (float*)alloc((size_t)cE * 4 * cDM * 4);
  float*  alp   = (float*)alloc((size_t)cNNZ * 4 * 4);
  float*  emax  = (float*)alloc((size_t)cE * 4 * 4);
  float*  esum  = (float*)alloc((size_t)cE * 4 * 4);
  float*  Dinv  = (float*)alloc((size_t)cN * 4);
  float*  Binv  = (float*)alloc((size_t)cE * 4);

  auto fill = [&](float* p, float v, long n) {
    k_fill<<<(int)cdivl(n, T), T, 0, stream>>>(p, v, n);
  };

  // ---- prep: bf16 activations + transposed bf16 weights ----
  k_f32_to_bf16<<<(int)cdivl((long)cN * cDIN, T), T, 0, stream>>>(x, x_bf, (long)cN * cDIN);
  k_transpose_bf16<<<(int)cdivl((long)cDIN * cDM, T), T, 0, stream>>>(W1, w1t, cDIN, cDM);
  k_transpose_bf16<<<(int)cdivl((long)cDM * 512, T), T, 0, stream>>>(h1_lin, h1lt, cDM, 512);
  k_transpose_bf16<<<(int)cdivl((long)cDM * cDM, T), T, 0, stream>>>(h2_lin, h2lt, cDM, cDM);
  k_transpose_bf16<<<(int)cdivl((long)cDM * cDIN, T), T, 0, stream>>>(W2, w2t, cDM, cDIN);

  // ---- degrees (shared by both hconvs) ----
  fill(Dinv, 0.f, cN); fill(Binv, 0.f, cE);
  k_count_deg<<<(int)cdivl(cNNZ, T), T, 0, stream>>>(vidx, eidx, Dinv, Binv, cNNZ);
  k_invert<<<(int)cdivl(cN, T), T, 0, stream>>>(Dinv, cN);
  k_invert<<<(int)cdivl(cE, T), T, 0, stream>>>(Binv, cE);

  // ---- h = leaky(x @ W1 + b1); bn1 -> hA (+ bf16 copy) ----
  launch_gemm(x_bf, w1t, b1, nullptr, hB, cN, cDM, cDIN, 1, stream);
  k_col_stats<<<cDM, 256, 0, stream>>>(hB, cN, cDM, mean, var);
  k_bn_apply<<<(int)cdivl((long)cN * cDM, T), T, 0, stream>>>(hB, mean, var, bn1g, bn1b,
                                                              hA, hbf, cN, cDM);

  // ---- hypergraph conv (shared sequence) ----
  auto hconv = [&](const __bf16* linT, int lin_out, const float* att, const float* bias,
                   int H, const float* bng, const float* bnb) {
    long nnzH = (long)cNNZ * H;
    long nnzHD = nnzH * 128;
    // he_attr = segment-sum of h over edges (== a.T @ h)
    fill(he, 0.f, (long)cE * cDM);
    k_scatter_he<<<(int)cdivl((long)cNNZ * cDM, T), T, 0, stream>>>(hA, vidx, eidx, he,
                                                                    (long)cNNZ * cDM);
    k_f32_to_bf16<<<(int)cdivl((long)cE * cDM, T), T, 0, stream>>>(he, hebf, (long)cE * cDM);
    // xh = h @ lin ; eh = he_attr @ lin   (WMMA)
    launch_gemm(hbf, linT, nullptr, nullptr, xh, cN, lin_out, cDM, 0, stream);
    launch_gemm(hebf, linT, nullptr, nullptr, eh, cE, lin_out, cDM, 0, stream);
    // attention logits + segment softmax over edges
    k_alpha<<<(int)cdivl(nnzH, T), T, 0, stream>>>(xh, eh, vidx, eidx, att, alp, H, nnzH);
    fill(emax, -1e30f, (long)cE * H);
    k_seg_max<<<(int)cdivl(nnzH, T), T, 0, stream>>>(alp, eidx, emax, H, nnzH);
    fill(esum, 0.f, (long)cE * H);
    k_seg_expsum<<<(int)cdivl(nnzH, T), T, 0, stream>>>(alp, eidx, emax, esum, H, nnzH);
    k_seg_norm<<<(int)cdivl(nnzH, T), T, 0, stream>>>(alp, eidx, esum, H, nnzH);
    // node->edge then edge->node message passing
    fill(oute, 0.f, (long)cE * H * 128);
    k_m1<<<(int)cdivl(nnzHD, T), T, 0, stream>>>(xh, alp, Binv, vidx, eidx, oute, H, nnzHD);
    fill(hC, 0.f, (long)cN * cDM);
    k_m2<<<(int)cdivl(nnzHD, T), T, 0, stream>>>(oute, alp, Dinv, vidx, eidx, hC,
                                                 H, 1.0f / H, nnzHD);
    // residual + bias, then BN back into hA / hbf
    k_add3<<<(int)cdivl((long)cN * cDM, T), T, 0, stream>>>(hA, hC, bias, hB,
                                                            (long)cN * cDM, cDM);
    k_col_stats<<<cDM, 256, 0, stream>>>(hB, cN, cDM, mean, var);
    k_bn_apply<<<(int)cdivl((long)cN * cDM, T), T, 0, stream>>>(hB, mean, var, bng, bnb,
                                                                hA, hbf, cN, cDM);
  };

  hconv(h1lt, 512, h1_att, h1_bias, 4, bn2g, bn2b);   // heads=4, bn2
  hconv(h2lt, 128, h2_att, h2_bias, 1, bn3g, bn3b);   // heads=1, bn3

  // ---- out = x + leaky(h @ W2 + b2)  (fused into d_out) ----
  launch_gemm(hbf, w2t, b2, x, out, cN, cDIN, cDM, 1, stream);

  // ---- bn4: per-channel over (N,P) -> view d_out as [N*P, DM], in place ----
  k_col_stats<<<cDM, 256, 0, stream>>>(out, cN * cP, cDM, mean, var);
  k_bn_apply<<<(int)cdivl((long)cN * cP * cDM, T), T, 0, stream>>>(out, mean, var, bn4g, bn4b,
                                                                   out, nullptr,
                                                                   (long)cN * cP, cDM);
}
